// SpeechEncDec_80118319939753
// MI455X (gfx1250) — compile-verified, matrix operand order checked by reading, not run on hardware
//
#include <hip/hip_runtime.h>
#include <hip/hip_bf16.h>

// ---------------------------------------------------------------------------
// MI455X / gfx1250 implementation.
//  - bf16 WMMA (v_wmma_f32_16x16x32_bf16) for both the per-layer input GEMM
//    and the recurrent GEMV (h broadcast across the 16 N-columns).
//  - W_hh is staged to bf16 once per layer (halves L2 traffic for the 256
//    re-reads during the scan).
//  - One persistent 16-wave (512-thread, wave32) workgroup runs all 512 GRU
//    layers; the sequential dependence chain syncs via s_barrier, not
//    kernel launches.
// ---------------------------------------------------------------------------

typedef __attribute__((ext_vector_type(16))) __bf16 bf16x16;
typedef __attribute__((ext_vector_type(8)))  __bf16 bf16x8;
typedef __attribute__((ext_vector_type(4)))  __bf16 bf16x4;
typedef __attribute__((ext_vector_type(8)))  float  f32x8;
typedef __attribute__((ext_vector_type(4)))  float  f32x4;

static __device__ __forceinline__ bf16x16 cat8(bf16x8 lo, bf16x8 hi) {
  return __builtin_shufflevector(lo, hi, 0,1,2,3,4,5,6,7,8,9,10,11,12,13,14,15);
}
static __device__ __forceinline__ bf16x8 cvt_f8(f32x8 f) {
  bf16x8 r;
#pragma unroll
  for (int i = 0; i < 8; ++i) r[i] = (__bf16)f[i];
  return r;
}
static __device__ __forceinline__ f32x8 wmma_bf16(bf16x16 a, bf16x16 b, f32x8 c) {
  // 8 args: (neg_a, A, neg_b, B, c_mod, C, reuse_a, reuse_b)
  return __builtin_amdgcn_wmma_f32_16x16x32_bf16(false, a, false, b, (short)0, c,
                                                 false, false);
}

// 16-bit operand fragment: rows r0..r0+15, K = k0..k0+31, row stride ld.
// Lane L holds row r0+(L&15); lanes 0-15: K {k0..k0+7, k0+16..k0+23};
// lanes 16-31: K {k0+8..k0+15, k0+24..k0+31}.  (ISA 7.12.2, 16-bit A 16x32.)
static __device__ __forceinline__ bf16x16
frag_bf16(const __bf16* __restrict__ base, int r0, int ld, int k0, int lane) {
  const __bf16* p = base + (size_t)(r0 + (lane & 15)) * ld + k0 + ((lane >> 4) << 3);
  bf16x8 c0 = *(const bf16x8*)p;
  bf16x8 c1 = *(const bf16x8*)(p + 16);
  return cat8(c0, c1);
}
static __device__ __forceinline__ bf16x16
frag_f32(const float* __restrict__ base, int r0, int ld, int k0, int lane) {
  const float* p = base + (size_t)(r0 + (lane & 15)) * ld + k0 + ((lane >> 4) << 3);
  f32x8 c0 = *(const f32x8*)p;
  f32x8 c1 = *(const f32x8*)(p + 16);
  return cat8(cvt_f8(c0), cvt_f8(c1));
}

// ---------------------------------------------------------------------------
// Conv front-end: seq[t][j] = (conv2_w @ (conv1_w @ X[:,2t] + b1) + b2)[j]
// One block per kept time column; output stored bf16 for the WMMA pipeline.
// ---------------------------------------------------------------------------
__global__ void __launch_bounds__(256)
conv_front(const float* __restrict__ X,  const float* __restrict__ w1,
           const float* __restrict__ b1, const float* __restrict__ w2,
           const float* __restrict__ b2, __bf16* __restrict__ seqg) {
  __shared__ float xcol[16];
  __shared__ float x1[128];
  const int t = blockIdx.x, tid = threadIdx.x;
  if (tid < 13) xcol[tid] = X[tid * 512 + 2 * t];
  __syncthreads();
  if (tid < 128) {
    float a = b1[tid];
#pragma unroll
    for (int k = 0; k < 13; ++k) a += w1[tid * 13 + k] * xcol[k];
    x1[tid] = a;
  }
  __syncthreads();
  float a = b2[tid];
  for (int k = 0; k < 128; ++k) a += w2[tid * 128 + k] * x1[k];
  seqg[t * 256 + tid] = (__bf16)a;
}

// ---------------------------------------------------------------------------
// Persistent GRU stack: 512 layers (256 enc + 256 dec, ReLU between stacks).
// ---------------------------------------------------------------------------
__global__ void __launch_bounds__(512)
gru_stack_kernel(const float* __restrict__ encWih, const float* __restrict__ encWhh,
                 const float* __restrict__ encbih, const float* __restrict__ encbhh,
                 const float* __restrict__ decWih, const float* __restrict__ decWhh,
                 const float* __restrict__ decbih, const float* __restrict__ decbhh,
                 float* __restrict__ xp, __bf16* __restrict__ whbf,
                 __bf16* __restrict__ seqg, float* __restrict__ encraw,
                 float* __restrict__ encrelu, float* __restrict__ decout) {
  __shared__ float hp[768];
  __shared__ float hcur[256];
  __shared__ __attribute__((aligned(16))) __bf16 hbf[256];
  __shared__ float bhs[768];

  const int tid = threadIdx.x, lane = tid & 31, wave = tid >> 5;

  for (int l = 0; l < 512; ++l) {
    const bool dec = l >= 256;
    const int  li  = l & 255;
    const float* Wi = (dec ? decWih : encWih) + (size_t)li * 768 * 256;
    const float* Wh = (dec ? decWhh : encWhh) + (size_t)li * 768 * 256;
    const float* bi = (dec ? decbih : encbih) + li * 768;
    const float* bh = (dec ? decbhh : encbhh) + li * 768;

    // ---- Phase 1: xp = seq @ Wi^T + bi  (M=256 time, N=768, K=256) ----
    for (int tile = wave; tile < 768; tile += 16) {
      const int m0 = (tile / 48) << 4;   // time tile
      const int n0 = (tile % 48) << 4;   // gate-feature tile
      f32x8 acc = {};
#pragma unroll
      for (int k0 = 0; k0 < 256; k0 += 32) {
        bf16x16 a = frag_bf16(seqg, m0, 256, k0, lane);       // activations
        bf16x16 b = frag_f32 (Wi,   n0, 256, k0, lane);       // Wi row n = B col n
        acc = wmma_bf16(a, b, acc);
      }
      // C/D layout: lane -> column, VGPR r -> row r (+8 for lanes 16-31)
      const int   col  = n0 + (lane & 15);
      const float bias = bi[col];
      const int   trow = m0 + ((lane >> 4) << 3);
#pragma unroll
      for (int r = 0; r < 8; ++r) xp[(trow + r) * 768 + col] = acc[r] + bias;
    }

    // ---- Phase 1b: stage Wh as bf16 (halves recurrent L2 read traffic) ----
    for (int i = tid; i < 768 * 256 / 4; i += 512) {
      f32x4 f = ((const f32x4*)Wh)[i];
      bf16x4 o;
#pragma unroll
      for (int e = 0; e < 4; ++e) o[e] = (__bf16)f[e];
      ((bf16x4*)whbf)[i] = o;
    }
    for (int i = tid; i < 768; i += 512) bhs[i] = bh[i];
    if (tid < 256) { hcur[tid] = 0.f; hbf[tid] = (__bf16)0.f; }
    __threadfence();
    __syncthreads();

    // ---- Phase 2: recurrent scan over T=256 ----
    for (int t = 0; t < 256; ++t) {
      // hp = Wh @ h : 48 row-tiles of 16, h broadcast across the 16 N-columns.
#pragma unroll
      for (int s = 0; s < 3; ++s) {
        const int m0 = (wave * 3 + s) << 4;
        f32x8 acc = {};
#pragma unroll
        for (int k0 = 0; k0 < 256; k0 += 32) {
          bf16x16 a = frag_bf16(whbf, m0, 256, k0, lane);
          const __bf16* p = &hbf[k0 + ((lane >> 4) << 3)];
          bf16x8 c0 = *(const bf16x8*)p;
          bf16x8 c1 = *(const bf16x8*)(p + 16);
          acc = wmma_bf16(a, cat8(c0, c1), acc);
        }
        if ((lane & 15) == 0) {           // all N-columns equal; lane 0/16 write
          const int r0 = m0 + ((lane >> 4) << 3);
#pragma unroll
          for (int r = 0; r < 8; ++r) hp[r0 + r] = acc[r];
        }
      }
      __syncthreads();
      if (tid < 256) {
        const int j = tid;
        const float* xpt = xp + t * 768;
        const float xr = xpt[j], xz = xpt[256 + j], xn = xpt[512 + j];
        const float hr = hp[j]       + bhs[j];
        const float hz = hp[256 + j] + bhs[256 + j];
        const float hn = hp[512 + j] + bhs[512 + j];
        const float r  = 1.f / (1.f + __expf(-(xr + hr)));
        const float z  = 1.f / (1.f + __expf(-(xz + hz)));
        const float n  = tanhf(xn + r * hn);
        const float hnew = (1.f - z) * n + z * hcur[j];
        hcur[j] = hnew;
        hbf[j]  = (__bf16)hnew;
        seqg[t * 256 + j] = (__bf16)hnew;     // input to next layer
        if (l == 255) encraw[t * 256 + j] = hnew;
        if (l == 511) decout[t * 256 + j] = hnew;
      }
      __syncthreads();
    }

    // ---- ReLU between the two stacks ----
    if (l == 255) {
      __threadfence();
      __syncthreads();
      for (int i = tid; i < 65536; i += 512) {
        const float v  = encraw[i];
        const float rl = v > 0.f ? v : 0.f;
        encrelu[i] = rl;
        seqg[i]    = (__bf16)rl;
      }
    }
    __threadfence();
    __syncthreads();
  }
}

// ---------------------------------------------------------------------------
// Attention + logits epilogue (faithful; the row-softmax over a 1-wide axis
// makes preds = exp(l - l)).
// ---------------------------------------------------------------------------
__global__ void __launch_bounds__(256)
attn_final(const float* __restrict__ enc, const float* __restrict__ dec,
           const float* __restrict__ fclw, const float* __restrict__ fclb,
           float* __restrict__ scores, float* __restrict__ out) {
  const int i = threadIdx.x;
  const float* di = dec + i * 256;
  float m = -3.402823466e38f;
  for (int j = 0; j < 256; ++j) {
    const float* ej = enc + j * 256;
    float s = 0.f;
    for (int k = 0; k < 256; ++k) s += di[k] * ej[k];
    scores[i * 256 + j] = s;
    m = fmaxf(m, s);
  }
  float denom = 0.f;
  for (int j = 0; j < 256; ++j) {
    const float e = __expf(scores[i * 256 + j] - m);
    scores[i * 256 + j] = e;
    denom += e;
  }
  const float inv = 1.f / denom;
  float acc_dec = 0.f;
  for (int k = 0; k < 256; ++k) acc_dec += fclw[k] * di[k];
  float acc_c = 0.f;
  const float a0 = scores[i * 256 + 0] * inv;
  for (int k = 0; k < 256; ++k) {
    float c = a0 * enc[k];                       // attn[:, :1] * enc[0]
    for (int j = 0; j < 255; ++j)                // attn[:, :S-1] @ enc[1:]
      c += (scores[i * 256 + j] * inv) * enc[(j + 1) * 256 + k];
    acc_c += fclw[256 + k] * c;
  }
  const float logit = acc_dec + acc_c + fclb[0];
  out[i] = __expf(logit - logit);                // softmax over 1-wide axis
}

// ---------------------------------------------------------------------------
extern "C" void kernel_launch(void* const* d_in, const int* in_sizes, int n_in,
                              void* d_out, int out_size, void* d_ws, size_t ws_size,
                              hipStream_t stream) {
  (void)in_sizes; (void)n_in; (void)out_size; (void)ws_size;
  const float* X   = (const float*)d_in[0];
  const float* w1  = (const float*)d_in[1];
  const float* b1  = (const float*)d_in[2];
  const float* w2  = (const float*)d_in[3];
  const float* b2  = (const float*)d_in[4];
  const float* eWi = (const float*)d_in[5];
  const float* eWh = (const float*)d_in[6];
  const float* ebi = (const float*)d_in[7];
  const float* ebh = (const float*)d_in[8];
  const float* dWi = (const float*)d_in[9];
  const float* dWh = (const float*)d_in[10];
  const float* dbi = (const float*)d_in[11];
  const float* dbh = (const float*)d_in[12];
  const float* fw  = (const float*)d_in[13];
  const float* fb  = (const float*)d_in[14];
  float* out = (float*)d_out;

  char* w = (char*)d_ws;
  float*  xp      = (float*)w;  w += 196608 * sizeof(float);   // [256,768]
  __bf16* whbf    = (__bf16*)w; w += 196608 * sizeof(__bf16);  // [768,256] bf16
  __bf16* seqg    = (__bf16*)w; w += 65536  * sizeof(__bf16);  // [256,256] bf16
  float*  encraw  = (float*)w;  w += 65536  * sizeof(float);
  float*  encrelu = (float*)w;  w += 65536  * sizeof(float);
  float*  decout  = (float*)w;  w += 65536  * sizeof(float);
  float*  scores  = (float*)w;  w += 65536  * sizeof(float);

  conv_front<<<256, 256, 0, stream>>>(X, w1, b1, w2, b2, seqg);
  gru_stack_kernel<<<1, 512, 0, stream>>>(eWi, eWh, ebi, ebh, dWi, dWh, dbi, dbh,
                                          xp, whbf, seqg, encraw, encrelu, decout);
  attn_final<<<1, 256, 0, stream>>>(encrelu, decout, fw, fb, scores, out);
}
